// MultiHeadAttention_34703335752346
// MI455X (gfx1250) — matrix operand order repaired and checked by reading, not compiled, not run on hardware
//
#include <hip/hip_runtime.h>
#include <hip/hip_bf16.h>

typedef __attribute__((ext_vector_type(16))) _Float16 v16h;
typedef __attribute__((ext_vector_type(8)))  _Float16 v8h;
typedef __attribute__((ext_vector_type(8)))  float    v8f;

#define B_  2
#define S_  2048
#define D_  768
#define H_  12
#define E_  64
#define M_  (B_*S_)   /* 4096 */
#define KSTEPS_ (D_/32)  /* 24 */

// ---------------------------------------------------------------------------
// WMMA fragment helpers (wave32; A/B use this code's contiguous k order, not the ISA table order)
// A (16-bit, 16x32 MxK): lane L -> row = L&15, hi = L>>4;
//   halves 0..15 = K [hi*16 .. hi*16+15]  (contiguous)
// B (16-bit, 32x16 KxN): lane L -> col = L&15;
//   halves 0..15 = K [ (L>>4)*16 .. (L>>4)*16+15 ]  (contiguous)
// C/D (f32 16x16): lane L -> col = L&15, VGPR r -> row = r + 8*(L>>4)
// ---------------------------------------------------------------------------

__device__ __forceinline__ v16h load_a_frag(const _Float16* base, int ld, int k0, int lane) {
  const int row = lane & 15;
  const int hi  = lane >> 4;
  const _Float16* rp = base + (size_t)row * ld + k0 + hi * 16;
  v8h lo = *(const v8h*)(rp);
  v8h hv = *(const v8h*)(rp + 8);
  v16h r;
#pragma unroll
  for (int i = 0; i < 8; ++i) { r[i] = lo[i]; r[i + 8] = hv[i]; }
  return r;
}

// bt is a transposed (N-major) matrix: bt[col][k]; returns frag for K block k0..k0+31
__device__ __forceinline__ v16h load_b_frag(const _Float16* bt, int ld, int k0, int lane) {
  const int col = lane & 15;
  const int kk  = k0 + ((lane >> 4) << 4);
  return *(const v16h*)(bt + (size_t)col * ld + kk);
}

__device__ __forceinline__ v8f wmma_f16(v16h a, v16h b, v8f c) {
  return __builtin_amdgcn_wmma_f32_16x16x32_f16(false, a, false, b, (short)0, c, false, false);
}

// Intra-wave LDS RAW fence: LDS ops from one wave complete in order (DScnt);
// this pins compiler ordering and drains outstanding DS ops.
__device__ __forceinline__ void lds_fence_wave() {
  asm volatile("s_wait_dscnt 0x0" ::: "memory");
}

// ---------------------------------------------------------------------------
// Conversion / transpose kernels
// ---------------------------------------------------------------------------
__global__ void k_f32_to_f16(const float* __restrict__ src, _Float16* __restrict__ dst, int n) {
  int i = blockIdx.x * blockDim.x + threadIdx.x;
  if (i < n) dst[i] = (_Float16)src[i];
}

// W [H][D][E] (fp32) -> Wt [H][E][D] (f16)
__global__ void k_transpose_whde(const float* __restrict__ w, _Float16* __restrict__ wt) {
  int i = blockIdx.x * blockDim.x + threadIdx.x;
  if (i >= H_ * D_ * E_) return;
  int e = i % E_;
  int d = (i / E_) % D_;
  int h = i / (D_ * E_);
  wt[((size_t)h * E_ + e) * D_ + d] = (_Float16)w[i];
}

// Wo [D][D] (fp32, k-major) -> Wot [n][k] (f16)
__global__ void k_transpose_wo(const float* __restrict__ w, _Float16* __restrict__ wt) {
  int i = blockIdx.x * blockDim.x + threadIdx.x;
  if (i >= D_ * D_) return;
  int n = i % D_;
  int k = i / D_;
  wt[(size_t)n * D_ + k] = (_Float16)w[i];
}

// ---------------------------------------------------------------------------
// Fully-unrolled ping-pong GEMM core: acc[4] += A[16 x 768] * Bt[4 tiles]
// Buffer parity is compile-time, so no register-rotation moves / hazard NOPs.
// ---------------------------------------------------------------------------
__device__ __forceinline__ void gemm_16x64_k768(
    const _Float16* __restrict__ abase, const _Float16* __restrict__ btbase,
    int bt_ld, int lane, v8f acc[4])
{
  v16h aF[2];
  v16h bF[2][4];
  aF[0] = load_a_frag(abase, D_, 0, lane);
#pragma unroll
  for (int nt = 0; nt < 4; ++nt)
    bF[0][nt] = load_b_frag(btbase + (size_t)(nt * 16) * bt_ld, bt_ld, 0, lane);

#pragma unroll
  for (int s = 0; s < KSTEPS_; ++s) {
    const int cur = s & 1;
    const int nxt = cur ^ 1;
    if (s + 1 < KSTEPS_) {
      const int kn = (s + 1) * 32;
      aF[nxt] = load_a_frag(abase, D_, kn, lane);
#pragma unroll
      for (int nt = 0; nt < 4; ++nt)
        bF[nxt][nt] = load_b_frag(btbase + (size_t)(nt * 16) * bt_ld, bt_ld, kn, lane);
    }
#pragma unroll
    for (int nt = 0; nt < 4; ++nt)
      acc[nt] = wmma_f16(aF[cur], bF[cur][nt], acc[nt]);
  }
}

// ---------------------------------------------------------------------------
// QKV projection: per head h, per 16-row M tile: [16 x 768] @ [768 x 64] + bias
// out layout: transposed_store==0 -> [H][B][S][E] (Q, K)
//             transposed_store==1 -> [H][B][E][S] (V, for contiguous PV B-frags)
// ---------------------------------------------------------------------------
__global__ __launch_bounds__(128) void k_qkv_gemm(
    const _Float16* __restrict__ xh,   // [M][D]
    const _Float16* __restrict__ wt,   // [H][E][D]
    const float*    __restrict__ bias, // [H][E]
    _Float16*       __restrict__ out,
    int transposed_store)
{
  const int lane  = threadIdx.x & 31;
  const int w     = threadIdx.x >> 5;
  const int mtile = blockIdx.x * 4 + w;   // [0,256)
  const int h     = blockIdx.y;
  const int m0    = mtile * 16;
  const int b     = m0 >> 11;
  const int s0    = m0 & (S_ - 1);
  const int col   = lane & 15;
  const int hi    = lane >> 4;

  v8f acc[4] = {};
  gemm_16x64_k768(xh + (size_t)m0 * D_, wt + (size_t)h * E_ * D_, D_, lane, acc);

#pragma unroll
  for (int nt = 0; nt < 4; ++nt) {
    const int e  = nt * 16 + col;
    const float bvv = bias[h * E_ + e];
    if (!transposed_store) {
      _Float16* ob = out + (((size_t)h * B_ + b) * S_ + s0) * E_;
#pragma unroll
      for (int r = 0; r < 8; ++r) {
        int row = r + 8 * hi;
        ob[(size_t)row * E_ + e] = (_Float16)(acc[nt][r] + bvv);
      }
    } else {
      _Float16* ob = out + (((size_t)h * B_ + b) * E_ + e) * S_ + s0;
#pragma unroll
      for (int r = 0; r < 8; ++r) {
        int row = r + 8 * hi;
        ob[row] = (_Float16)(acc[nt][r] + bvv);
      }
    }
  }
}

// ---------------------------------------------------------------------------
// Flash attention: one wave per (h, b, 16-query-row tile); online softmax over
// 2048 keys in 32-key blocks. 4 WMMA for logits + 4 WMMA for PV per block.
// K and V fragments + mask (NT) are all issued at the top of the iteration so
// the softmax tail covers the V-fragment latency before the PV WMMAs.
// ---------------------------------------------------------------------------
__global__ __launch_bounds__(128) void k_attention(
    const _Float16* __restrict__ Qf,   // [H][B][S][E]
    const _Float16* __restrict__ Kf,   // [H][B][S][E]
    const _Float16* __restrict__ Vt,   // [H][B][E][S]
    const float*    __restrict__ mask, // [H*B][S][S]
    _Float16*       __restrict__ ctx)  // [B][S][H*E]
{
  __shared__ _Float16 lds_p[4][16 * 32];

  const int lane = threadIdx.x & 31;
  const int w    = threadIdx.x >> 5;
  const int job  = blockIdx.x * 4 + w;     // [0, 3072)
  const int qt   = job & 127;
  const int hb   = job >> 7;               // h*2 + b
  const int h    = hb >> 1;
  const int b    = hb & 1;
  const int s0   = qt * 16;
  const int col  = lane & 15;
  const int hi   = lane >> 4;

  const _Float16* Qhb  = Qf + (size_t)hb * S_ * E_;
  const _Float16* Khb  = Kf + (size_t)hb * S_ * E_;
  const _Float16* Vhb  = Vt + (size_t)hb * E_ * S_;
  const float*    mrow = mask + ((size_t)hb * S_ + s0) * S_;

  v16h aq0 = load_a_frag(Qhb + (size_t)s0 * E_, E_, 0,  lane);
  v16h aq1 = load_a_frag(Qhb + (size_t)s0 * E_, E_, 32, lane);

  v8f o0 = {}, o1 = {}, o2 = {}, o3 = {};
  float mrun[8], lrun[8];
#pragma unroll
  for (int r = 0; r < 8; ++r) { mrun[r] = -3.0e38f; lrun[r] = 0.0f; }

  for (int j = 0; j < S_ / 32; ++j) {
    const int t0 = j * 32;
    if (j + 1 < S_ / 32) {
      __builtin_prefetch(Khb + (size_t)(t0 + 32) * E_, 0, 3);
      __builtin_prefetch(Vhb + t0 + 32, 0, 3);
    }

    // issue ALL global loads for this block up front: K frags, V frags, mask
    v16h bk0 = load_b_frag(Khb + (size_t)t0 * E_,        E_, 0,  lane);
    v16h bk1 = load_b_frag(Khb + (size_t)t0 * E_,        E_, 32, lane);
    v16h bk2 = load_b_frag(Khb + (size_t)(t0 + 16) * E_, E_, 0,  lane);
    v16h bk3 = load_b_frag(Khb + (size_t)(t0 + 16) * E_, E_, 32, lane);
    v16h bv0 = load_b_frag(Vhb + t0,                   S_, 0, lane);
    v16h bv1 = load_b_frag(Vhb + (size_t)16 * S_ + t0, S_, 0, lane);
    v16h bv2 = load_b_frag(Vhb + (size_t)32 * S_ + t0, S_, 0, lane);
    v16h bv3 = load_b_frag(Vhb + (size_t)48 * S_ + t0, S_, 0, lane);

    float mk0[8], mk1[8];
#pragma unroll
    for (int r = 0; r < 8; ++r) {
      int row = r + 8 * hi;
      const float* mp = mrow + (size_t)row * S_ + t0;
      mk0[r] = __builtin_nontemporal_load(mp + col);
      mk1[r] = __builtin_nontemporal_load(mp + 16 + col);
    }

    // logits tile [16 x 32] = Q[16x64] @ K[t0..t0+31]^T
    v8f c0 = {}, c1 = {};
    c0 = wmma_f16(aq0, bk0, c0);
    c0 = wmma_f16(aq1, bk1, c0);
    c1 = wmma_f16(aq0, bk2, c1);
    c1 = wmma_f16(aq1, bk3, c1);

    // additive mask
#pragma unroll
    for (int r = 0; r < 8; ++r) { c0[r] += mk0[r]; c1[r] += mk1[r]; }

    // online softmax (row reductions across the 16-lane half holding each row)
    float mnew[8], alpha[8];
#pragma unroll
    for (int r = 0; r < 8; ++r) {
      float mx = fmaxf(c0[r], c1[r]);
      mx = fmaxf(mx, __shfl_xor(mx, 1, 32));
      mx = fmaxf(mx, __shfl_xor(mx, 2, 32));
      mx = fmaxf(mx, __shfl_xor(mx, 4, 32));
      mx = fmaxf(mx, __shfl_xor(mx, 8, 32));
      mnew[r]  = fmaxf(mrun[r], mx);
      alpha[r] = __expf(mrun[r] - mnew[r]);
      mrun[r]  = mnew[r];
    }
#pragma unroll
    for (int r = 0; r < 8; ++r) {
      float p0 = __expf(c0[r] - mnew[r]);
      float p1 = __expf(c1[r] - mnew[r]);
      c0[r] = p0; c1[r] = p1;
      float rs = p0 + p1;
      rs += __shfl_xor(rs, 1, 32);
      rs += __shfl_xor(rs, 2, 32);
      rs += __shfl_xor(rs, 4, 32);
      rs += __shfl_xor(rs, 8, 32);
      lrun[r] = lrun[r] * alpha[r] + rs;
      o0[r] *= alpha[r]; o1[r] *= alpha[r]; o2[r] *= alpha[r]; o3[r] *= alpha[r];
    }

    // C-frag -> A-frag transpose through LDS (strictly intra-wave tile;
    // DS ops from one wave complete in order, so no block barrier needed)
    _Float16* lp = lds_p[w];
#pragma unroll
    for (int r = 0; r < 8; ++r) {
      int row = r + 8 * hi;
      lp[row * 32 + col]      = (_Float16)c0[r];
      lp[row * 32 + 16 + col] = (_Float16)c1[r];
    }
    lds_fence_wave();
    v16h ap = load_a_frag(lp, 32, 0, lane);

    // O += P[16x32] @ V[t0..t0+31][0..63]   (V frags already resident)
    o0 = wmma_f16(ap, bv0, o0);
    o1 = wmma_f16(ap, bv1, o1);
    o2 = wmma_f16(ap, bv2, o2);
    o3 = wmma_f16(ap, bv3, o3);
    lds_fence_wave();   // P tile fully consumed before next iteration rewrites it
  }

  // normalize and store into concat-head context (f16)
  _Float16* cb = ctx + ((size_t)b * S_ + s0) * D_ + h * E_;
#pragma unroll
  for (int r = 0; r < 8; ++r) {
    int row = r + 8 * hi;
    float inv = 1.0f / lrun[r];
    cb[(size_t)row * D_ + 0  + col] = (_Float16)(o0[r] * inv);
    cb[(size_t)row * D_ + 16 + col] = (_Float16)(o1[r] * inv);
    cb[(size_t)row * D_ + 32 + col] = (_Float16)(o2[r] * inv);
    cb[(size_t)row * D_ + 48 + col] = (_Float16)(o3[r] * inv);
  }
}

// ---------------------------------------------------------------------------
// Output projection: out[4096 x 768] = ctx @ Wo + bo (fp32 out, NT stores)
// ---------------------------------------------------------------------------
__global__ __launch_bounds__(128) void k_out_proj(
    const _Float16* __restrict__ ctx,  // [M][D]
    const _Float16* __restrict__ wot,  // [n][k] = Wo^T
    const float*    __restrict__ bo,   // [D]
    float*          __restrict__ out)  // [M][D]
{
  const int lane  = threadIdx.x & 31;
  const int w     = threadIdx.x >> 5;
  const int mtile = blockIdx.x * 4 + w;
  const int m0    = mtile * 16;
  const int n0    = blockIdx.y * 64;
  const int col   = lane & 15;
  const int hi    = lane >> 4;

  v8f acc[4] = {};
  gemm_16x64_k768(ctx + (size_t)m0 * D_, wot + (size_t)n0 * D_, D_, lane, acc);

#pragma unroll
  for (int nt = 0; nt < 4; ++nt) {
    int n = n0 + nt * 16 + col;
    float bb = bo[n];
#pragma unroll
    for (int r = 0; r < 8; ++r) {
      int row = m0 + r + 8 * hi;
      __builtin_nontemporal_store(acc[nt][r] + bb, out + (size_t)row * D_ + n);
    }
  }
}

// ---------------------------------------------------------------------------
// Launch
// ---------------------------------------------------------------------------
extern "C" void kernel_launch(void* const* d_in, const int* in_sizes, int n_in,
                              void* d_out, int out_size, void* d_ws, size_t ws_size,
                              hipStream_t stream) {
  (void)in_sizes; (void)n_in; (void)out_size; (void)ws_size;

  const float* x   = (const float*)d_in[0];
  const float* msk = (const float*)d_in[1];
  const float* Wq  = (const float*)d_in[2];
  const float* bq  = (const float*)d_in[3];
  const float* Wk  = (const float*)d_in[4];
  const float* bk  = (const float*)d_in[5];
  const float* Wv  = (const float*)d_in[6];
  const float* bv  = (const float*)d_in[7];
  const float* Wo  = (const float*)d_in[8];
  const float* bo  = (const float*)d_in[9];
  float* out = (float*)d_out;

  char* ws = (char*)d_ws;
  size_t off = 0;
  auto take = [&](size_t bytes) -> void* {
    void* p = ws + off;
    off += (bytes + 255) & ~(size_t)255;
    return p;
  };

  _Float16* xh  = (_Float16*)take((size_t)M_ * D_ * 2);
  _Float16* wqt = (_Float16*)take((size_t)H_ * E_ * D_ * 2);
  _Float16* wkt = (_Float16*)take((size_t)H_ * E_ * D_ * 2);
  _Float16* wvt = (_Float16*)take((size_t)H_ * E_ * D_ * 2);
  _Float16* wot = (_Float16*)take((size_t)D_ * D_ * 2);
  _Float16* Qf  = (_Float16*)take((size_t)H_ * B_ * S_ * E_ * 2);
  _Float16* Kf  = (_Float16*)take((size_t)H_ * B_ * S_ * E_ * 2);
  _Float16* Vtp = (_Float16*)take((size_t)H_ * B_ * E_ * S_ * 2);
  _Float16* ctx = (_Float16*)take((size_t)M_ * D_ * 2);

  int n;
  n = M_ * D_;
  k_f32_to_f16<<<(n + 255) / 256, 256, 0, stream>>>(x, xh, n);
  n = H_ * D_ * E_;
  k_transpose_whde<<<(n + 255) / 256, 256, 0, stream>>>(Wq, wqt);
  k_transpose_whde<<<(n + 255) / 256, 256, 0, stream>>>(Wk, wkt);
  k_transpose_whde<<<(n + 255) / 256, 256, 0, stream>>>(Wv, wvt);
  n = D_ * D_;
  k_transpose_wo<<<(n + 255) / 256, 256, 0, stream>>>(Wo, wot);

  dim3 gq(64, 12);
  k_qkv_gemm<<<gq, 128, 0, stream>>>(xh, wqt, bq, Qf, 0);
  k_qkv_gemm<<<gq, 128, 0, stream>>>(xh, wkt, bk, Kf, 0);
  k_qkv_gemm<<<gq, 128, 0, stream>>>(xh, wvt, bv, Vtp, 1);

  k_attention<<<768, 128, 0, stream>>>(Qf, Kf, Vtp, msk, ctx);

  k_out_proj<<<dim3(64, 12), 128, 0, stream>>>(ctx, wot, bo, out);
}
